// TriangularUpdateTE_71090298683425
// MI455X (gfx1250) — matrix operand order repaired and checked by reading, not compiled, not run on hardware
//
#include <hip/hip_runtime.h>
#include <hip/hip_bf16.h>
#include <math.h>

// ---------------------------------------------------------------------------
// TriangularUpdate for MI455X (gfx1250): all GEMM-shaped work routed through
// v_wmma_f32_16x16x32_bf16 (f32 accumulate). LayerNorm / sigmoid / mask in f32.
// Workspace layout (requires ~134.4 MB):
//   [0      ,  32768) WpiT  bf16[128][128]  (pi_w transposed)
//   [32768  ,  65536) WgiT  bf16[128][128]
//   [65536  ,  81920) WpoT  bf16[128][64]
//   [81920  ,  98304) WgoT  bf16[128][64]
//   [131072 , +32MB ) Abuf  bf16[64][512][512]  planes 0-31: a1[i][k], 32-63: a2^T[i][k]
//   [.. +32MB, +32MB) Bbuf  bf16[64][512][512]  planes 0-31: b1[j][k], 32-63: b2^T[j][k]
//   [.. +64MB, +64MB) tb    f32 [64][512][512]  planes 0-31: x1,       32-63: x2
// ---------------------------------------------------------------------------

typedef __bf16 v16bf __attribute__((ext_vector_type(16)));
typedef float  v8f   __attribute__((ext_vector_type(8)));

union FragU { uint4 q[2]; v16bf v; };
union Pack8 { unsigned short us[8]; uint4 q; };

__device__ __forceinline__ unsigned short f2bf(float f) {
    unsigned u = __float_as_uint(f);
    u += 0x7FFFu + ((u >> 16) & 1u);          // round-to-nearest-even
    return (unsigned short)(u >> 16);
}
// fast sigmoid: v_exp_f32 + v_rcp_f32 (avoids the precise-divide expansion)
__device__ __forceinline__ float sigm(float x) {
    return __builtin_amdgcn_rcpf(1.0f + __expf(-x));
}

// ---------------------------------------------------------------------------
// Kernel 0: transpose weights (in,out)->(out,in), convert to bf16.
// ---------------------------------------------------------------------------
__global__ __launch_bounds__(256) void prep_weights(
    const float* __restrict__ pi_w, const float* __restrict__ gi_w,
    const float* __restrict__ po_w, const float* __restrict__ go_w,
    unsigned short* __restrict__ WpiT, unsigned short* __restrict__ WgiT,
    unsigned short* __restrict__ WpoT, unsigned short* __restrict__ WgoT)
{
    const int t = threadIdx.x;
    for (int idx = t; idx < 128 * 128; idx += 256) {
        const int o = idx >> 7, k = idx & 127;
        WpiT[o * 128 + k] = f2bf(pi_w[k * 128 + o]);
        WgiT[o * 128 + k] = f2bf(gi_w[k * 128 + o]);
    }
    for (int idx = t; idx < 128 * 64; idx += 256) {
        const int o = idx & 127, k = idx >> 7;       // k<64, o<128
        WpoT[o * 64 + k] = f2bf(po_w[k * 128 + o]);
        WgoT[o * 64 + k] = f2bf(go_w[k * 128 + o]);
    }
}

// ---------------------------------------------------------------------------
// Kernel 1: LN(128) -> pi/gi WMMA GEMMs -> sigmoid gate -> mask -> planar bf16
// Block = 16x16 (i,j) tile of the 512x512 grid, 256 threads (8 waves).
// ---------------------------------------------------------------------------
__global__ __launch_bounds__(256) void k1_ln_gate(
    const float* __restrict__ x, const float* __restrict__ mask,
    const float* __restrict__ ln1w, const float* __restrict__ ln1b,
    const float* __restrict__ pib, const float* __restrict__ gib,
    const unsigned short* __restrict__ WpiT,
    const unsigned short* __restrict__ WgiT,
    unsigned short* __restrict__ Abuf, unsigned short* __restrict__ Bbuf)
{
    __shared__ unsigned short hbf[256 * 136];   // normalized rows, bf16, padded stride
    __shared__ float msk[256];

    const int t  = threadIdx.x;
    const int I0 = blockIdx.y * 16, J0 = blockIdx.x * 16;
    const int li = t >> 4, lj = t & 15;
    const size_t rowflat = (size_t)(I0 + li) * 512 + (size_t)(J0 + lj);
    const float4* xr = (const float4*)(x + rowflat * 128);

    // ---- LayerNorm over D=128 (two streaming passes over the row) ----
    float s = 0.f, s2 = 0.f;
    for (int k4 = 0; k4 < 32; ++k4) {
        const float4 v = xr[k4];
        s  += v.x + v.y + v.z + v.w;
        s2 += v.x * v.x + v.y * v.y + v.z * v.z + v.w * v.w;
    }
    const float mu   = s * (1.f / 128.f);
    const float var  = s2 * (1.f / 128.f) - mu * mu;
    const float rstd = rsqrtf(var + 1e-5f);

    unsigned short* ha = &hbf[t * 136];
    for (int k4 = 0; k4 < 32; ++k4) {
        const float4 v = xr[k4];
        const float vv[4] = { v.x, v.y, v.z, v.w };
        for (int c = 0; c < 4; ++c) {
            const int k = k4 * 4 + c;
            ha[k] = f2bf((vv[c] - mu) * rstd * ln1w[k] + ln1b[k]);
        }
    }
    msk[t] = mask[rowflat];
    __syncthreads();

    // ---- WMMA GEMMs + gating ----
    const int w = t >> 5, lane = t & 31, lr = lane & 15, sel = lane >> 4;

    for (int half = 0; half < 2; ++half) {          // 0: i-fixed tiles (ch 0-63), 1: j-fixed (ch 64-127)
        for (int e = 0; e < 2; ++e) {
            const int tile = 2 * w + e;             // 0..15
            const int arow = half ? (lr * 16 + tile) : (tile * 16 + lr);
            const unsigned short* ap = &hbf[arow * 136];
            FragU a[4];
            for (int kt = 0; kt < 4; ++kt) {
                const int o0 = kt * 32 + sel * 8;   // ISA A layout: K 0-7/16-23 (lo lanes), 8-15/24-31 (hi)
                a[kt].q[0] = *(const uint4*)(ap + o0);
                a[kt].q[1] = *(const uint4*)(ap + o0 + 16);
            }
            for (int nt2 = 0; nt2 < 4; ++nt2) {
                const int cg    = nt2 * 16 + lr;    // channel within half: 0..63
                const int cfull = half * 64 + cg;   // full channel 0..127
                v8f accP = {}, accG = {};
                const unsigned short* wp = WpiT + (size_t)cfull * 128;
                const unsigned short* wg = WgiT + (size_t)cfull * 128;
                for (int kt = 0; kt < 4; ++kt) {
                    FragU bp, bg;
                    const int kb = kt * 32 + sel * 8;   // B layout matches A: K 0-7/16-23 (lo lanes), 8-15/24-31 (hi)
                    bp.q[0] = *(const uint4*)(wp + kb);
                    bp.q[1] = *(const uint4*)(wp + kb + 16);
                    bg.q[0] = *(const uint4*)(wg + kb);
                    bg.q[1] = *(const uint4*)(wg + kb + 16);
                    accP = __builtin_amdgcn_wmma_f32_16x16x32_bf16(false, a[kt].v, false, bp.v,
                                                                   (short)0, accP, false, false);
                    accG = __builtin_amdgcn_wmma_f32_16x16x32_bf16(false, a[kt].v, false, bg.v,
                                                                   (short)0, accG, false, false);
                }
                const float biasP = pib[cfull];
                const float biasG = gib[cfull];
                Pack8 pk;
                for (int v = 0; v < 8; ++v) {
                    const int m    = v + 8 * sel;
                    const int trow = half ? (m * 16 + tile) : (tile * 16 + m);
                    const float g  = (accP[v] + biasP) * sigm(accG[v] + biasG) * msk[trow];
                    pk.us[v] = f2bf(g);
                }
                unsigned short* dst = (cg < 32) ? Abuf : Bbuf;
                const int plane = (cg & 31) + (half ? 32 : 0);
                const int prow  = (half ? J0 : I0) + tile;
                const int pcol  = (half ? I0 : J0) + 8 * sel;
                *(uint4*)(dst + (size_t)plane * 262144 + (size_t)prow * 512 + pcol) = pk.q;
            }
        }
    }
}

// ---------------------------------------------------------------------------
// Kernel 2: per-plane C[i][j] = sum_k P[i][k] * Q[j][k]  (both einsums).
// Block = one d-plane, 64x64 output tile; K staged through LDS in 64-chunks
// (8 chunks, 4 WMMAs per wave between barriers).
// ---------------------------------------------------------------------------
__global__ __launch_bounds__(256) void k2_einsum(
    const unsigned short* __restrict__ Abuf,
    const unsigned short* __restrict__ Bbuf,
    float* __restrict__ tb)
{
    __shared__ unsigned short Pa[64 * 72];   // 64 rows x 64 K, padded stride 72
    __shared__ unsigned short Qa[64 * 72];

    const int t  = threadIdx.x;
    const int d  = blockIdx.z;
    const int I0 = blockIdx.y * 64, J0 = blockIdx.x * 64;
    const unsigned short* P = Abuf + (size_t)d * 262144;
    const unsigned short* Q = Bbuf + (size_t)d * 262144;

    const int w = t >> 5, lane = t & 31, lr = lane & 15, sel = lane >> 4;
    const int ti  = w >> 1;            // i-tile (0..3)
    const int tj0 = (w & 1) * 2;       // first of two j-tiles
    const int lrow  = t >> 2;          // cooperative loader: row 0..63
    const int lpart = (t & 3) * 16;    // 32B piece within 128B row chunk

    v8f acc0 = {}, acc1 = {};
    for (int kc = 0; kc < 8; ++kc) {
        const int k0 = kc * 64;
        __syncthreads();
        {
            const size_t gp = (size_t)(I0 + lrow) * 512 + k0 + lpart;
            const size_t gq = (size_t)(J0 + lrow) * 512 + k0 + lpart;
            *(uint4*)&Pa[lrow * 72 + lpart]     = *(const uint4*)&P[gp];
            *(uint4*)&Pa[lrow * 72 + lpart + 8] = *(const uint4*)&P[gp + 8];
            *(uint4*)&Qa[lrow * 72 + lpart]     = *(const uint4*)&Q[gq];
            *(uint4*)&Qa[lrow * 72 + lpart + 8] = *(const uint4*)&Q[gq + 8];
        }
        __syncthreads();

        const unsigned short* ar  = &Pa[(ti * 16 + lr) * 72];
        const unsigned short* br0 = &Qa[(tj0 * 16 + lr) * 72];
        const unsigned short* br1 = &Qa[((tj0 + 1) * 16 + lr) * 72];
        for (int kt = 0; kt < 2; ++kt) {
            FragU a, b0, b1;
            const int oa = kt * 32 + sel * 8;
            a.q[0] = *(const uint4*)(ar + oa);
            a.q[1] = *(const uint4*)(ar + oa + 16);
            const int ob = kt * 32 + sel * 8;
            b0.q[0] = *(const uint4*)(br0 + ob);
            b0.q[1] = *(const uint4*)(br0 + ob + 16);
            b1.q[0] = *(const uint4*)(br1 + ob);
            b1.q[1] = *(const uint4*)(br1 + ob + 16);
            acc0 = __builtin_amdgcn_wmma_f32_16x16x32_bf16(false, a.v, false, b0.v, (short)0, acc0, false, false);
            acc1 = __builtin_amdgcn_wmma_f32_16x16x32_bf16(false, a.v, false, b1.v, (short)0, acc1, false, false);
        }
    }

    float* tp = tb + (size_t)d * 262144;
    for (int v = 0; v < 8; ++v) {
        const int i = I0 + ti * 16 + v + 8 * sel;
        tp[(size_t)i * 512 + J0 + tj0 * 16 + lr]       = acc0[v];
        tp[(size_t)i * 512 + J0 + (tj0 + 1) * 16 + lr] = acc1[v];
    }
}

// ---------------------------------------------------------------------------
// Kernel 3: LN(64) -> po/go WMMA GEMMs (K=64) -> sigmoid gate -> mask -> out.
// Block = 16x16 (i,j) tile, 256 threads.
// ---------------------------------------------------------------------------
__global__ __launch_bounds__(256) void k3_out(
    const float* __restrict__ tb, const float* __restrict__ mask,
    const float* __restrict__ ln2w, const float* __restrict__ ln2b,
    const float* __restrict__ pob, const float* __restrict__ gob,
    const unsigned short* __restrict__ WpoT,
    const unsigned short* __restrict__ WgoT,
    float* __restrict__ out)
{
    __shared__ float          ts[256 * 65];     // t rows gathered from planar, stride 65
    __shared__ unsigned short h2bf[256 * 72];   // normalized rows bf16, stride 72
    __shared__ float          msk[256];

    const int t  = threadIdx.x;
    const int I0 = blockIdx.y * 16, J0 = blockIdx.x * 16;
    const int li = t >> 4, lj = t & 15;
    const size_t gpos = (size_t)(I0 + li) * 512 + (size_t)(J0 + lj);

    for (int d = 0; d < 64; ++d)
        ts[t * 65 + d] = tb[(size_t)d * 262144 + gpos];
    msk[t] = mask[gpos];

    float s = 0.f, s2 = 0.f;
    for (int k = 0; k < 64; ++k) { const float v = ts[t * 65 + k]; s += v; s2 += v * v; }
    const float mu   = s * (1.f / 64.f);
    const float rstd = rsqrtf(s2 * (1.f / 64.f) - mu * mu + 1e-5f);
    for (int k = 0; k < 64; ++k)
        h2bf[t * 72 + k] = f2bf((ts[t * 65 + k] - mu) * rstd * ln2w[k] + ln2b[k]);
    __syncthreads();

    const int w = t >> 5, lane = t & 31, lr = lane & 15, sel = lane >> 4;
    for (int e = 0; e < 2; ++e) {
        const int tile = 2 * w + e;
        const unsigned short* ap = &h2bf[(tile * 16 + lr) * 72];
        FragU a[2];
        for (int kt = 0; kt < 2; ++kt) {
            a[kt].q[0] = *(const uint4*)(ap + kt * 32 + sel * 8);
            a[kt].q[1] = *(const uint4*)(ap + kt * 32 + 16 + sel * 8);
        }
        for (int nt = 0; nt < 8; ++nt) {
            const int c = nt * 16 + lr;
            v8f accP = {}, accG = {};
            const unsigned short* wp = WpoT + (size_t)c * 64;
            const unsigned short* wg = WgoT + (size_t)c * 64;
            for (int kt = 0; kt < 2; ++kt) {
                FragU bp, bg;
                const int kb = kt * 32 + sel * 8;
                bp.q[0] = *(const uint4*)(wp + kb);
                bp.q[1] = *(const uint4*)(wp + kb + 16);
                bg.q[0] = *(const uint4*)(wg + kb);
                bg.q[1] = *(const uint4*)(wg + kb + 16);
                accP = __builtin_amdgcn_wmma_f32_16x16x32_bf16(false, a[kt].v, false, bp.v,
                                                               (short)0, accP, false, false);
                accG = __builtin_amdgcn_wmma_f32_16x16x32_bf16(false, a[kt].v, false, bg.v,
                                                               (short)0, accG, false, false);
            }
            const float biasP = pob[c];
            const float biasG = gob[c];
            for (int v = 0; v < 8; ++v) {
                const int m    = v + 8 * sel;
                const int trow = tile * 16 + m;
                const float r  = (accP[v] + biasP) * sigm(accG[v] + biasG) * msk[trow];
                const size_t orow = (size_t)(I0 + tile) * 512 + (size_t)(J0 + m);
                out[orow * 128 + c] = r;
            }
        }
    }
}

// ---------------------------------------------------------------------------
extern "C" void kernel_launch(void* const* d_in, const int* in_sizes, int n_in,
                              void* d_out, int out_size, void* d_ws, size_t ws_size,
                              hipStream_t stream) {
    const float* x    = (const float*)d_in[0];
    const float* mask = (const float*)d_in[1];
    const float* ln1w = (const float*)d_in[2];
    const float* ln1b = (const float*)d_in[3];
    const float* pi_w = (const float*)d_in[4];
    const float* pi_b = (const float*)d_in[5];
    const float* gi_w = (const float*)d_in[6];
    const float* gi_b = (const float*)d_in[7];
    const float* ln2w = (const float*)d_in[8];
    const float* ln2b = (const float*)d_in[9];
    const float* po_w = (const float*)d_in[10];
    const float* po_b = (const float*)d_in[11];
    const float* go_w = (const float*)d_in[12];
    const float* go_b = (const float*)d_in[13];
    float* out = (float*)d_out;

    char* ws = (char*)d_ws;
    unsigned short* WpiT = (unsigned short*)(ws + 0);
    unsigned short* WgiT = (unsigned short*)(ws + 32768);
    unsigned short* WpoT = (unsigned short*)(ws + 65536);
    unsigned short* WgoT = (unsigned short*)(ws + 81920);
    unsigned short* Abuf = (unsigned short*)(ws + 131072);
    unsigned short* Bbuf = (unsigned short*)(ws + 131072 + 33554432ull);
    float*          tb   = (float*)        (ws + 131072 + 67108864ull);
    // total workspace needed: 131072 + 2*32MB + 64MB = 134,348,800 bytes

    prep_weights<<<1, 256, 0, stream>>>(pi_w, gi_w, po_w, go_w, WpiT, WgiT, WpoT, WgoT);
    k1_ln_gate<<<dim3(32, 32), 256, 0, stream>>>(x, mask, ln1w, ln1b, pi_b, gi_b,
                                                 WpiT, WgiT, Abuf, Bbuf);
    k2_einsum<<<dim3(8, 8, 64), 256, 0, stream>>>(Abuf, Bbuf, tb);
    k3_out<<<dim3(32, 32), 256, 0, stream>>>(tb, mask, ln2w, ln2b, po_b, go_b,
                                             WpoT, WgoT, out);
}
